// GlobalAttentionRepViTBlock_4226247819438
// MI455X (gfx1250) — matrix operand order repaired and checked by reading, not compiled, not run on hardware
//
#include <hip/hip_runtime.h>
#include <hip/hip_bf16.h>
#include <math.h>

// ---------------------------------------------------------------------------
// GlobalAttentionRepViT block for MI455X (gfx1250), wave32 + WMMA bf16.
// Pipeline: dwconv -> bn1 stats -> bn1+SE-pool -> SE MLP -> scale+pack ->
//           QKV GEMMs (WMMA) -> flash attention (WMMA) + residual ->
//           pw GEMM -> bn2 -> m1 GEMM -> bn3+GELU -> m2 GEMM -> bn4 -> out
// Activations feeding GEMMs are token-major [N,K] so every WMMA operand is
// two 16-byte b128 loads per lane (no scalar gathers / sub-word packing).
// ---------------------------------------------------------------------------

typedef __attribute__((ext_vector_type(16))) __bf16 v16bf;
typedef __attribute__((ext_vector_type(8)))  __bf16 bf16x8;
typedef __attribute__((ext_vector_type(8)))  float  v8f;

#define BSZ 8
#define Cc  128
#define HH  56
#define NN  3136      /* 56*56 */
#define OO  256
#define HD  512

static __device__ __forceinline__ v8f wmma_bf16(v16bf a, v16bf b, v8f c) {
  // v_wmma_f32_16x16x32_bf16  D = A(16x32) * B(32x16) + C(16x16 f32)
  return __builtin_amdgcn_wmma_f32_16x16x32_bf16(false, a, false, b,
                                                 (short)0, c, false, false);
}

// Build a 16-element bf16 operand from two aligned 16-byte chunks.
template <typename P>
static __device__ __forceinline__ v16bf load16(P p0, P p1) {
  union { v16bf v; bf16x8 h[2]; } u;
  u.h[0] = *(const bf16x8*)p0;
  u.h[1] = *(const bf16x8*)p1;
  return u.v;
}

// --------------------------- depthwise conv 3x3 s2 -------------------------
__global__ void dwconv_s2(const float* __restrict__ x, const float* __restrict__ w,
                          float* __restrict__ h)
{
  long i = (long)blockIdx.x * blockDim.x + threadIdx.x;
  if (i >= (long)BSZ * Cc * NN) return;
  int ow = (int)(i % HH);
  int oh = (int)((i / HH) % HH);
  int c  = (int)((i / NN) % Cc);
  int b  = (int)(i / ((long)NN * Cc));
  const float* wp = w + c * 9;
  const float* xp = x + ((long)(b * Cc + c)) * 112 * 112;
  float acc = 0.f;
#pragma unroll
  for (int kh = 0; kh < 3; ++kh) {
    int ih = 2 * oh - 1 + kh;
    if (ih < 0 || ih >= 112) continue;
#pragma unroll
    for (int kw = 0; kw < 3; ++kw) {
      int iw = 2 * ow - 1 + kw;
      if (iw < 0 || iw >= 112) continue;
      acc += xp[ih * 112 + iw] * wp[kh * 3 + kw];
    }
  }
  h[i] = acc;
}

// ----------------- per-channel BN stats (training-mode, biased var) --------
// Y layout: [B][M][NN] f32; one block per channel m.
__global__ __launch_bounds__(256) void bnstats(
    const float* __restrict__ Y, const float* __restrict__ g,
    const float* __restrict__ beta, float* __restrict__ scale,
    float* __restrict__ shift, int M)
{
  __shared__ float rs[256], rq[256];
  int m = blockIdx.x;
  float s = 0.f, q = 0.f;
  for (int b = 0; b < BSZ; ++b) {
    const float* row = Y + ((long)b * M + m) * NN;
    for (int n = threadIdx.x; n < NN; n += 256) {
      float v = row[n];
      s += v; q += v * v;
    }
  }
  rs[threadIdx.x] = s; rq[threadIdx.x] = q;
  __syncthreads();
  for (int st = 128; st > 0; st >>= 1) {
    if (threadIdx.x < st) {
      rs[threadIdx.x] += rs[threadIdx.x + st];
      rq[threadIdx.x] += rq[threadIdx.x + st];
    }
    __syncthreads();
  }
  if (threadIdx.x == 0) {
    float inv  = 1.f / (float)(BSZ * NN);
    float mean = rs[0] * inv;
    float var  = rq[0] * inv - mean * mean;
    float sc   = g[m] * rsqrtf(var + 1e-5f);
    scale[m] = sc;
    shift[m] = beta[m] - mean * sc;
  }
}

// ------------- bn1 apply (in place) + SE global average pool ---------------
__global__ __launch_bounds__(256) void bn1_se_pool(
    float* __restrict__ h, const float* __restrict__ scale,
    const float* __restrict__ shift, float* __restrict__ sepool)
{
  __shared__ float red[256];
  int bc = blockIdx.x;
  int c  = bc & (Cc - 1);
  float sc = scale[c], sh = shift[c];
  float* p = h + (long)bc * NN;
  float s = 0.f;
  for (int n = threadIdx.x; n < NN; n += 256) {
    float v = p[n] * sc + sh;
    p[n] = v;
    s += v;
  }
  red[threadIdx.x] = s; __syncthreads();
  for (int st = 128; st > 0; st >>= 1) {
    if (threadIdx.x < st) red[threadIdx.x] += red[threadIdx.x + st];
    __syncthreads();
  }
  if (threadIdx.x == 0) sepool[bc] = red[0] * (1.f / (float)NN);
}

// ----------------------------- SE MLP (tiny) -------------------------------
__global__ __launch_bounds__(256) void se_mlp(
    const float* __restrict__ sepool, const float* __restrict__ w1,
    const float* __restrict__ b1, const float* __restrict__ w2,
    const float* __restrict__ b2, float* __restrict__ sescale)
{
  __shared__ float s1[BSZ * 32];
  int t = threadIdx.x;
  {
    int b = t >> 5, r = t & 31;
    float a = b1[r];
    for (int c = 0; c < Cc; ++c) a += sepool[b * Cc + c] * w1[r * Cc + c];
    s1[t] = fmaxf(a, 0.f);
  }
  __syncthreads();
  for (int idx = t; idx < BSZ * Cc; idx += 256) {
    int b = idx >> 7, c = idx & 127;
    float a = b2[c];
    for (int r = 0; r < 32; ++r) a += s1[b * 32 + r] * w2[c * 32 + r];
    sescale[idx] = 1.f / (1.f + __expf(-a));
  }
}

// -- SE scale (in place f32 [B][C][N]) + token-major bf16 pack [B][N][C] ----
__global__ __launch_bounds__(256) void se_scale_pack(
    float* __restrict__ h, const float* __restrict__ sescale,
    __bf16* __restrict__ hfb)
{
  int bc = blockIdx.x;
  int b = bc >> 7, c = bc & 127;
  float s = sescale[bc];
  float* p = h + (long)bc * NN;
  __bf16* o = hfb + (long)b * NN * Cc + c;
  for (int n = threadIdx.x; n < NN; n += 256) {
    float v = p[n] * s;
    p[n] = v;
    o[(long)n * Cc] = (__bf16)v;
  }
}

// --------------------------- f32 -> bf16 convert ---------------------------
__global__ void cvt_f32_bf16(const float* __restrict__ s, __bf16* __restrict__ d, int n)
{
  int i = blockIdx.x * blockDim.x + threadIdx.x;
  if (i < n) d[i] = (__bf16)s[i];
}

// ---------------------------------------------------------------------------
// WMMA GEMM: out[m,n] = sum_k W[m,k]*X[n,k] (+bias[m]), per batch z.
//   W: [M,K] bf16 row-major (shared across batches)
//   X: [N,K] bf16 token-major, batch stride xBatch elements
//   MODE 0: f32 out row-major [M,NN]; 1: bf16 out row-major [M,NN];
//   MODE 2: bf16 out transposed out[n*32+m] (builds Q/K as [N,32] padded).
// Block: 256 threads = 8 waves; wave tile 16x16; block tile 64x32.
// ---------------------------------------------------------------------------
template <int MODE, bool HAS_BIAS>
__global__ __launch_bounds__(256) void gemm_wmma(
    const __bf16* __restrict__ W, const __bf16* __restrict__ X,
    const float* __restrict__ bias, void* __restrict__ out,
    int M, int K, long xBatch, long oBatch)
{
  int lane = threadIdx.x & 31;
  int wave = threadIdx.x >> 5;
  int wm = wave >> 1, wn = wave & 1;
  int m0 = blockIdx.x * 64 + wm * 16;
  int n0 = blockIdx.y * 32 + wn * 16;
  if (m0 >= M) return;
  int bz = blockIdx.z;

  int col  = lane & 15;
  int half = lane >> 4;
  int kbA = half * 16;     // A K base per lane half
  int kbB = half * 16;     // B K base per lane half
  const __bf16* Arow = W + (long)(m0 + col) * K;
  const __bf16* Xrow = X + (long)bz * xBatch + (long)(n0 + col) * K;
  int n = n0 + col;

  v8f acc = {0.f, 0.f, 0.f, 0.f, 0.f, 0.f, 0.f, 0.f};
  for (int k0 = 0; k0 < K; k0 += 32) {
    v16bf a = load16(Arow + k0 + kbA, Arow + k0 + kbA + 8);
    v16bf b = load16(Xrow + k0 + kbB, Xrow + k0 + kbB + 8);
    __builtin_prefetch(Arow + k0 + 32, 0, 0);  // global_prefetch_b8
    acc = wmma_bf16(a, b, acc);
  }

  int rbase = half * 8;
#pragma unroll
  for (int j = 0; j < 8; ++j) {
    int m = m0 + rbase + j;
    float v = acc[j];
    if (HAS_BIAS) v += bias[m];
    if (MODE == 0)
      ((float*)out)[(long)bz * oBatch + (long)m * NN + n] = v;
    else if (MODE == 1)
      ((__bf16*)out)[(long)bz * oBatch + (long)m * NN + n] = (__bf16)v;
    else
      ((__bf16*)out)[(long)bz * oBatch + (long)n * 32 + m] = (__bf16)v;
  }
}

// ---------------------------------------------------------------------------
// Flash attention: S = Q Kt (no scaling), row softmax over all 3136 keys,
// O[m,c] = sum_n P[m,n] V[c,n]; epilogue divides by row-sum, adds residual h,
// writes bf16 x2 token-major [n][c] (ready as pw GEMM operand).
// Block = 4 waves; each wave owns one 16-query tile; streams 32 keys/step.
// ---------------------------------------------------------------------------
__global__ __launch_bounds__(128) void flash_attn(
    const __bf16* __restrict__ Qnd, const __bf16* __restrict__ Knd,
    const __bf16* __restrict__ V, const float* __restrict__ hres,
    __bf16* __restrict__ x2)
{
  __shared__ float  sS[4][16][32];
  __shared__ __bf16 sP[4][16][32];
  __shared__ float  sM[4][16], sL[4][16], sF[4][16];

  int lane = threadIdx.x & 31;
  int w    = threadIdx.x >> 5;
  int b    = blockIdx.y;
  int m0   = (blockIdx.x * 4 + w) * 16;

  const __bf16* Qb = Qnd + (long)b * NN * 32;
  const __bf16* Kb = Knd + (long)b * NN * 32;
  const __bf16* Vb = V    + (long)b * Cc * NN;
  const float*  Hb = hres + (long)b * Cc * NN;
  __bf16*       Ob = x2   + (long)b * NN * Cc;

  int col = lane & 15, half = lane >> 4;
  int kbA = half * 16, kbB = half * 16, rbase = half * 8;

  if (lane < 16) { sM[w][lane] = -3.0e38f; sL[w][lane] = 0.f; }

  // A operand: Q tile (16 queries x 32 padded d), constant over the loop
  const __bf16* qrow = Qb + (long)(m0 + col) * 32;
  v16bf aQ = load16(qrow + kbA, qrow + kbA + 8);

  v8f acc[8];
  v8f z = {0.f, 0.f, 0.f, 0.f, 0.f, 0.f, 0.f, 0.f};
#pragma unroll
  for (int t = 0; t < 8; ++t) acc[t] = z;
  __syncthreads();

  for (int n0 = 0; n0 < NN; n0 += 32) {
    // ---- S tiles: two 16x16 WMMAs (K'=32, d rows 16..31 are zero pad) ----
    const __bf16* krow  = Kb + (long)(n0 + col) * 32;
    const __bf16* krow2 = Kb + (long)(n0 + 16 + col) * 32;
    v8f s0 = wmma_bf16(aQ, load16(krow  + kbB, krow  + kbB + 8), z);
    v8f s1 = wmma_bf16(aQ, load16(krow2 + kbB, krow2 + kbB + 8), z);
#pragma unroll
    for (int j = 0; j < 8; ++j) {
      sS[w][rbase + j][col]      = s0[j];
      sS[w][rbase + j][16 + col] = s1[j];
    }
    __syncthreads();

    // ---- online softmax (one lane per query row) ----
    if (lane < 16) {
      float mold = sM[w][lane], mnew = mold;
#pragma unroll
      for (int c = 0; c < 32; ++c) mnew = fmaxf(mnew, sS[w][lane][c]);
      float f  = __expf(mold - mnew);
      float ls = sL[w][lane] * f;
#pragma unroll
      for (int c = 0; c < 32; ++c) {
        float e = __expf(sS[w][lane][c] - mnew);
        sP[w][lane][c] = (__bf16)e;
        ls += e;
      }
      sM[w][lane] = mnew; sL[w][lane] = ls; sF[w][lane] = f;
    }
    __syncthreads();

    // ---- rescale O accumulators by exp(m_old - m_new) per row ----
#pragma unroll
    for (int j = 0; j < 8; ++j) {
      float f = sF[w][rbase + j];
#pragma unroll
      for (int t = 0; t < 8; ++t) acc[t][j] *= f;
    }

    // ---- A operand: P tile from LDS (two ds_load_b128) ----
    const __bf16* prow = &sP[w][col][0];
    v16bf aP = load16(prow + kbA, prow + kbA + 8);

    // ---- O += P(16x32) x Vt(32x16) for 8 channel tiles ----
#pragma unroll
    for (int t = 0; t < 8; ++t) {
      int ch = t * 16 + col;
      const __bf16* vrow = Vb + (long)ch * NN + n0;
      acc[t] = wmma_bf16(aP, load16(vrow + kbB, vrow + kbB + 8), acc[t]);
    }
    __syncthreads();
  }

  // ---- epilogue: /row-sum, add residual h, write bf16 token-major ----
#pragma unroll
  for (int t = 0; t < 8; ++t) {
    int ch = t * 16 + col;
#pragma unroll
    for (int j = 0; j < 8; ++j) {
      int q = m0 + rbase + j;
      float o = acc[t][j] / sL[w][rbase + j];
      Ob[(long)q * Cc + ch] = (__bf16)(o + Hb[(long)ch * NN + q]);
    }
  }
}

// -------- BN apply (+optional exact GELU); one (b,m) row per block ---------
// In: Y f32 [B][M][NN]. Out: TOKENOUT ? bf16 [B][NN][M] : f32 [B][M][NN].
template <bool GELU, bool TOKENOUT>
__global__ __launch_bounds__(256) void bn_apply(
    const float* __restrict__ Y, const float* __restrict__ scale,
    const float* __restrict__ shift, void* __restrict__ out, int M)
{
  int bm = blockIdx.x;
  int b = bm / M, m = bm - b * M;
  float sc = scale[m], sh = shift[m];
  const float* row = Y + (long)bm * NN;
  for (int n = threadIdx.x; n < NN; n += 256) {
    float v = row[n] * sc + sh;
    if (GELU) v = 0.5f * v * (1.f + erff(v * 0.70710678118f));
    if (TOKENOUT)
      ((__bf16*)out)[((long)b * NN + n) * M + m] = (__bf16)v;
    else
      ((float*)out)[(long)bm * NN + n] = v;
  }
}

// ---------------------------------------------------------------------------
extern "C" void kernel_launch(void* const* d_in, const int* in_sizes, int n_in,
                              void* d_out, int out_size, void* d_ws, size_t ws_size,
                              hipStream_t stream)
{
  const float* x     = (const float*)d_in[0];
  const float* dw_w  = (const float*)d_in[1];
  const float* bn1_g = (const float*)d_in[2];
  const float* bn1_b = (const float*)d_in[3];
  const float* se_w1 = (const float*)d_in[4];
  const float* se_b1 = (const float*)d_in[5];
  const float* se_w2 = (const float*)d_in[6];
  const float* se_b2 = (const float*)d_in[7];
  const float* q_w   = (const float*)d_in[8];
  const float* q_b   = (const float*)d_in[9];
  const float* k_w   = (const float*)d_in[10];
  const float* k_b   = (const float*)d_in[11];
  const float* v_w   = (const float*)d_in[12];
  const float* v_b   = (const float*)d_in[13];
  const float* pw_w  = (const float*)d_in[14];
  const float* bn2_g = (const float*)d_in[15];
  const float* bn2_b = (const float*)d_in[16];
  const float* m1_w  = (const float*)d_in[17];
  const float* bn3_g = (const float*)d_in[18];
  const float* bn3_b = (const float*)d_in[19];
  const float* m2_w  = (const float*)d_in[20];
  const float* bn4_g = (const float*)d_in[21];
  const float* bn4_b = (const float*)d_in[22];
  (void)in_sizes; (void)n_in; (void)out_size; (void)ws_size;

  char* ws = (char*)d_ws;

  // --------- workspace plan (with lifetime-based reuse, ~100 MB) ----------
  constexpr size_t SZ_H   = (size_t)BSZ * Cc * NN * 4;   // h f32 (in-place reused)
  constexpr size_t SZ_HFB = (size_t)BSZ * Cc * NN * 2;   // bf16 [B,N,128]
  constexpr size_t SZ_QK  = (size_t)BSZ * NN * 32 * 2;   // bf16 [B,N,32] padded
  constexpr size_t SZ_PW  = (size_t)BSZ * OO * NN * 4;
  constexpr size_t SZ_X3  = (size_t)BSZ * OO * NN * 2;
  constexpr size_t SZ_X4  = (size_t)BSZ * HD * NN * 2;

  const size_t OFF_H   = 0;
  const size_t OFF_HFB = OFF_H + SZ_H;
  const size_t OFF_Q   = OFF_HFB + SZ_HFB;
  const size_t OFF_K   = OFF_Q + SZ_QK;
  const size_t OFF_V   = OFF_K + SZ_QK;
  const size_t OFF_X2  = OFF_V + SZ_HFB;
  const size_t OFF_PW  = OFF_X2 + SZ_HFB;
  const size_t OFF_X3  = OFF_PW + SZ_PW;          // 61,014,016
  const size_t OFF_M1  = 0;                       // reuses [0, OFF_X3): dead by then
  const size_t OFF_X4  = OFF_X3 + SZ_X3;          // 73,859,072
  const size_t OFF_M2  = 0;                       // reuses [0, ...): dead by then
  size_t cur = OFF_X4 + SZ_X4;                    // smalls @ 99,549,184
  auto take = [&](size_t bytes) {
    size_t o = cur; cur += (bytes + 255) & ~(size_t)255; return o;
  };

  float* bn1s = (float*)(ws + take(Cc * 4));
  float* bn1h = (float*)(ws + take(Cc * 4));
  float* sepool  = (float*)(ws + take(BSZ * Cc * 4));
  float* sescale = (float*)(ws + take(BSZ * Cc * 4));
  float* bn2s = (float*)(ws + take(OO * 4));
  float* bn2h = (float*)(ws + take(OO * 4));
  float* bn3s = (float*)(ws + take(HD * 4));
  float* bn3h = (float*)(ws + take(HD * 4));
  float* bn4s = (float*)(ws + take(OO * 4));
  float* bn4h = (float*)(ws + take(OO * 4));
  __bf16* WQ  = (__bf16*)(ws + take(16 * Cc * 2));
  __bf16* WK  = (__bf16*)(ws + take(16 * Cc * 2));
  __bf16* WV  = (__bf16*)(ws + take(Cc * Cc * 2));
  __bf16* WPW = (__bf16*)(ws + take(OO * Cc * 2));
  __bf16* WM1 = (__bf16*)(ws + take(HD * OO * 2));
  __bf16* WM2 = (__bf16*)(ws + take(OO * HD * 2));

  float*  hF   = (float*)(ws + OFF_H);
  __bf16* hfB  = (__bf16*)(ws + OFF_HFB);   // [B][N][128]
  __bf16* Qnd  = (__bf16*)(ws + OFF_Q);     // [B][N][32]
  __bf16* Knd  = (__bf16*)(ws + OFF_K);     // [B][N][32]
  __bf16* Vbf  = (__bf16*)(ws + OFF_V);     // [B][128][N]
  __bf16* x2B  = (__bf16*)(ws + OFF_X2);    // [B][N][128]
  float*  pwF  = (float*)(ws + OFF_PW);     // [B][256][N]
  __bf16* x3B  = (__bf16*)(ws + OFF_X3);    // [B][N][256]
  float*  m1F  = (float*)(ws + OFF_M1);     // [B][512][N]
  __bf16* x4B  = (__bf16*)(ws + OFF_X4);    // [B][N][512]
  float*  m2F  = (float*)(ws + OFF_M2);     // [B][256][N]

  const long totCN = (long)BSZ * Cc * NN;   // 3,211,264

  // zero Q/K pads (d = 16..31 stay zero; GEMM writes d = 0..15)
  hipMemsetAsync(Qnd, 0, SZ_QK, stream);
  hipMemsetAsync(Knd, 0, SZ_QK, stream);

  // weight packs f32 -> bf16
  auto cvt = [&](const float* s, __bf16* d, int n) {
    cvt_f32_bf16<<<(n + 255) / 256, 256, 0, stream>>>(s, d, n);
  };
  cvt(q_w, WQ, 16 * Cc);  cvt(k_w, WK, 16 * Cc);  cvt(v_w, WV, Cc * Cc);
  cvt(pw_w, WPW, OO * Cc); cvt(m1_w, WM1, HD * OO); cvt(m2_w, WM2, OO * HD);

  // 1) depthwise conv
  dwconv_s2<<<(unsigned)((totCN + 255) / 256), 256, 0, stream>>>(x, dw_w, hF);
  // 2) bn1 stats + apply (in place) + SE pool
  bnstats<<<Cc, 256, 0, stream>>>(hF, bn1_g, bn1_b, bn1s, bn1h, Cc);
  bn1_se_pool<<<BSZ * Cc, 256, 0, stream>>>(hF, bn1s, bn1h, sepool);
  // 3) SE MLP + scale + token-major bf16 pack
  se_mlp<<<1, 256, 0, stream>>>(sepool, se_w1, se_b1, se_w2, se_b2, sescale);
  se_scale_pack<<<BSZ * Cc, 256, 0, stream>>>(hF, sescale, hfB);

  // 4) Q/K/V projections (WMMA GEMM); Q/K stored transposed [N,32] (padded)
  dim3 blk(256);
  dim3 gQ(1, NN / 32, BSZ);
  gemm_wmma<2, true><<<gQ, blk, 0, stream>>>(WQ, hfB, q_b, Qnd, 16, Cc,
                                             (long)NN * Cc, (long)NN * 32);
  gemm_wmma<2, true><<<gQ, blk, 0, stream>>>(WK, hfB, k_b, Knd, 16, Cc,
                                             (long)NN * Cc, (long)NN * 32);
  dim3 gV(Cc / 64, NN / 32, BSZ);
  gemm_wmma<1, true><<<gV, blk, 0, stream>>>(WV, hfB, v_b, Vbf, Cc, Cc,
                                             (long)NN * Cc, (long)Cc * NN);

  // 5) flash attention + residual -> x2 bf16 [B,N,128]
  dim3 gA(NN / 16 / 4, BSZ);     // 49 x 8, 4 query tiles per block
  flash_attn<<<gA, 128, 0, stream>>>(Qnd, Knd, Vbf, hF, x2B);

  // 6) pw 1x1 + bn2 -> x3 bf16 [B,N,256]
  dim3 gPW(OO / 64, NN / 32, BSZ);
  gemm_wmma<0, false><<<gPW, blk, 0, stream>>>(WPW, x2B, nullptr, pwF, OO, Cc,
                                               (long)NN * Cc, (long)OO * NN);
  bnstats<<<OO, 256, 0, stream>>>(pwF, bn2_g, bn2_b, bn2s, bn2h, OO);
  bn_apply<false, true><<<BSZ * OO, 256, 0, stream>>>(pwF, bn2s, bn2h, x3B, OO);

  // 7) mixer fc1 + bn3 + GELU -> x4 bf16 [B,N,512]
  dim3 gM1(HD / 64, NN / 32, BSZ);
  gemm_wmma<0, false><<<gM1, blk, 0, stream>>>(WM1, x3B, nullptr, m1F, HD, OO,
                                               (long)NN * OO, (long)HD * NN);
  bnstats<<<HD, 256, 0, stream>>>(m1F, bn3_g, bn3_b, bn3s, bn3h, HD);
  bn_apply<true, true><<<BSZ * HD, 256, 0, stream>>>(m1F, bn3s, bn3h, x4B, HD);

  // 8) mixer fc2 + bn4 -> out f32 [B,256,56,56]
  dim3 gM2(OO / 64, NN / 32, BSZ);
  gemm_wmma<0, false><<<gM2, blk, 0, stream>>>(WM2, x4B, nullptr, m2F, OO, HD,
                                               (long)NN * HD, (long)OO * NN);
  bnstats<<<OO, 256, 0, stream>>>(m2F, bn4_g, bn4_b, bn4s, bn4h, OO);
  bn_apply<false, false><<<BSZ * OO, 256, 0, stream>>>(m2F, bn4s, bn4h, d_out, OO);
}
